// MultiDense_51556787422041
// MI455X (gfx1250) — matrix operand order repaired and checked by reading, not compiled, not run on hardware
//
#include <hip/hip_runtime.h>

#define BSZ    2048
#define NSPLIT 256
#define OUTD   256
#define IND    256
#define ROWSTR (NSPLIT * IND)   // 65536: element stride between batch rows (x and out)

#define MT 128     // batch rows per block
#define NT 128     // out cols per block
#define KT 32      // k-chunk (bf16 WMMA K)
#define NK (IND / KT)
#define XSTR 40    // padded LDS row stride in halfwords (80B: 16B-aligned rows, no bank conflicts)

typedef __attribute__((ext_vector_type(16))) __bf16        v16bf;
typedef __attribute__((ext_vector_type(8)))  float         v8f;
typedef __attribute__((ext_vector_type(4)))  unsigned int  v4u;
typedef __attribute__((ext_vector_type(2)))  unsigned int  v2u;

union Frag { v16bf bf; v4u u[2]; };

// 4x fp32 bits -> 4x bf16 (round-half-up), packed via v_perm_b32: 3 VALU per pair
__device__ __forceinline__ v2u pack4(v4u v) {
  unsigned int ux = v.x + 0x8000u;
  unsigned int uy = v.y + 0x8000u;
  unsigned int uz = v.z + 0x8000u;
  unsigned int uw = v.w + 0x8000u;
  v2u o;
  o.x = __builtin_amdgcn_perm(uy, ux, 0x07060302u);  // {uy[31:16], ux[31:16]}
  o.y = __builtin_amdgcn_perm(uw, uz, 0x07060302u);
  return o;
}

__device__ __forceinline__ v4u ld16(const unsigned int* p) {
  return *(const v4u*)__builtin_assume_aligned(p, 16);
}

__global__ __launch_bounds__(256)
void multidense_bf16_wmma(const float* __restrict__ xf,
                          const float* __restrict__ Af,
                          const float* __restrict__ Bp,
                          float* __restrict__ out) {
  __shared__ alignas(16) unsigned short Xs[2][MT * XSTR];
  __shared__ alignas(16) unsigned short As[2][NT * XSTR];

  const unsigned int* x = (const unsigned int*)xf;
  const unsigned int* A = (const unsigned int*)Af;

  const int tid  = threadIdx.x;
  const int lane = tid & 31;
  const int wave = tid >> 5;
  const int h    = lane >> 4;   // lane half (K-split per ISA layout)
  const int lm   = lane & 15;   // M (A-frag) / N (B-frag) index within 16x16 tile

  const int blk = blockIdx.x;
  const int n   = blk >> 5;            // group; 32 adjacent blocks share A_n -> L2 resident
  const int rem = blk & 31;
  const int b0  = (rem >> 1) * MT;
  const int o0  = (rem & 1) * NT;

  const int wm = wave & 3;   // 4 wave-rows of 32 batch rows
  const int wn = wave >> 2;  // 2 wave-cols of 64 out cols

  // uniform-base + 32-bit offset addressing (all offsets < 2^27 elements)
  const int row0  = tid >> 3;          // row (X) / col (A) handled by this thread, j=0
  const int kq    = (tid & 7) << 2;    // k sub-offset (4 floats)
  const int xbase = (b0 + row0) * ROWSTR + n * IND + kq;
  const int abase = n * (OUTD * IND) + (o0 + row0) * IND + kq;
  const int lbase = row0 * XSTR + kq;  // LDS halfword index (same shape for Xs/As)

  v4u xr[4], ar[4];

#define LOAD_TILES(k0)                                                     \
  {                                                                        \
    _Pragma("unroll")                                                      \
    for (int j = 0; j < 4; ++j)                                            \
      xr[j] = ld16(x + xbase + j * (32 * ROWSTR) + (k0));                  \
    _Pragma("unroll")                                                      \
    for (int j = 0; j < 4; ++j)                                            \
      ar[j] = ld16(A + abase + j * (32 * IND) + (k0));                     \
  }

#define STASH_TILES(buf)                                                   \
  {                                                                        \
    _Pragma("unroll")                                                      \
    for (int j = 0; j < 4; ++j)                                            \
      *(v2u*)&Xs[buf][lbase + j * (32 * XSTR)] = pack4(xr[j]);             \
    _Pragma("unroll")                                                      \
    for (int j = 0; j < 4; ++j)                                            \
      *(v2u*)&As[buf][lbase + j * (32 * XSTR)] = pack4(ar[j]);             \
  }

  v8f acc[2][4];
  const v8f vz = {};
#pragma unroll
  for (int mt = 0; mt < 2; ++mt)
#pragma unroll
    for (int t = 0; t < 4; ++t) acc[mt][t] = vz;

  LOAD_TILES(0);
  STASH_TILES(0);
  __syncthreads();

#pragma unroll
  for (int kt = 0; kt < NK; ++kt) {
    const int cur = kt & 1;
    if (kt + 1 < NK) LOAD_TILES((kt + 1) * KT);

    // A fragments: lane half h covers K = 8h..8h+7 (u[0]) and 16+8h..16+8h+7 (u[1])
    Frag af[2];
#pragma unroll
    for (int mt = 0; mt < 2; ++mt) {
      const unsigned short* xrow = &Xs[cur][(wm * 32 + mt * 16 + lm) * XSTR];
      af[mt].u[0] = *(const v4u*)(xrow + 8 * h);
      af[mt].u[1] = *(const v4u*)(xrow + 16 + 8 * h);
    }

#pragma unroll
    for (int t = 0; t < 4; ++t) {
      // B fragment: column = lm, lane half h covers K = 8h..8h+7 (u[0]) and 16+8h..16+8h+7 (u[1])
      Frag bf;
      const unsigned short* arow = &As[cur][(wn * 64 + t * 16 + lm) * XSTR];
      bf.u[0] = *(const v4u*)(arow + 8 * h);
      bf.u[1] = *(const v4u*)(arow + 16 + 8 * h);
#pragma unroll
      for (int mt = 0; mt < 2; ++mt) {
        acc[mt][t] = __builtin_amdgcn_wmma_f32_16x16x32_bf16(
            false, af[mt].bf, false, bf.bf, (short)0, acc[mt][t], false, false);
      }
    }

    if (kt + 1 < NK) {
      STASH_TILES((kt + 1) & 1);   // other buffer; prior reads fenced by last barrier
      __syncthreads();
    }
  }

  // Epilogue: D layout — lane lm = column, VGPR r holds M = r + 8*h; bias = Bp[n, o]
  const int obase = (b0 + wm * 32 + 8 * h) * ROWSTR + n * OUTD + o0 + wn * 64 + lm;
  const int bbase = n * IND + o0 + wn * 64 + lm;
#pragma unroll
  for (int t = 0; t < 4; ++t) {
    const float bv = Bp[bbase + t * 16];
#pragma unroll
    for (int mt = 0; mt < 2; ++mt) {
#pragma unroll
      for (int r = 0; r < 8; ++r) {
        out[obase + t * 16 + (mt * 16 + r) * ROWSTR] = acc[mt][t][r] + bv;
      }
    }
  }
#undef LOAD_TILES
#undef STASH_TILES
}

extern "C" void kernel_launch(void* const* d_in, const int* in_sizes, int n_in,
                              void* d_out, int out_size, void* d_ws, size_t ws_size,
                              hipStream_t stream) {
  (void)in_sizes; (void)n_in; (void)out_size; (void)d_ws; (void)ws_size;
  const float* x  = (const float*)d_in[0];
  const float* A  = (const float*)d_in[1];
  const float* Bp = (const float*)d_in[2];
  float* out = (float*)d_out;

  dim3 grid(NSPLIT * (BSZ / MT) * (OUTD / NT));  // 256 * 16 * 2 = 8192 blocks
  dim3 block(256);                               // 8 wave32s
  multidense_bf16_wmma<<<grid, block, 0, stream>>>(x, A, Bp, out);
}
